// Head_3444563771822
// MI455X (gfx1250) — matrix operand order repaired and checked by reading, not compiled, not run on hardware
//
#include <hip/hip_runtime.h>
#include <hip/hip_bf16.h>

// MI455X / gfx1250 fused causal attention head.
//   ref: k = X@Wk, q = X@Wq, v = X@Wv;  S[i,j] = (k_i . q_j) * H^-0.5 (j<=i)
//        out = softmax(S) @ v
// Flash roles:  fQ = X@Wk,  fK = X@(Wq*scale),  fVt = (X@Wv)^T per batch.
// All matmuls via v_wmma_f32_16x16x32_f16 (fp32 accumulation).
// Global->LDS staging uses GLOBAL_LOAD_ASYNC_TO_LDS_B128 (ASYNCcnt-tracked
// DMA, no VGPR round trip) with a plain vector-copy fallback.

typedef __attribute__((ext_vector_type(16))) _Float16 v16h;
typedef __attribute__((ext_vector_type(8)))  _Float16 v8h;
typedef __attribute__((ext_vector_type(8)))  float    v8f;
typedef int v4i __attribute__((vector_size(16)));    // builtin's expected type

#define NB      8
#define TT      2048
#define CC      1024
#define HH      64
#define ROWS    (NB * TT)        // 16384 flattened (b,t) rows
#define SCALE   0.125f           // 64^-0.5

#if defined(__HIP_DEVICE_COMPILE__) && \
    __has_builtin(__builtin_amdgcn_global_load_async_to_lds_b128)
#define HAVE_ASYNC_LDS 1
#else
#define HAVE_ASYNC_LDS 0
#endif

// copy 16 bytes global -> LDS (async DMA if available, else through VGPRs)
static __device__ __forceinline__ void g2l16(const void* g, void* l) {
#if HAVE_ASYNC_LDS
    __builtin_amdgcn_global_load_async_to_lds_b128(
        (__attribute__((address_space(1))) v4i*)g,
        (__attribute__((address_space(3))) v4i*)l, 0, 0);
#else
    *(v8h*)l = *(const v8h*)g;
#endif
}

static __device__ __forceinline__ void async_wait() {
#if HAVE_ASYNC_LDS
#if __has_builtin(__builtin_amdgcn_s_wait_asynccnt)
    __builtin_amdgcn_s_wait_asynccnt(0);
#else
    asm volatile("s_wait_asynccnt 0x0" ::: "memory");
#endif
#endif
}

// ---------------------------------------------------------------------------
// Kernel 0: pack W^T (f32 [C][H] -> f16 [H][C]), fold softmax scale into Wq.
// grid = (3*H*C/256), block = 256.
// ---------------------------------------------------------------------------
__global__ __launch_bounds__(256) void head_pack_w(
    const float* __restrict__ Wk, const float* __restrict__ Wq,
    const float* __restrict__ Wv, _Float16* __restrict__ Wt)
{
    const int idx = blockIdx.x * 256 + threadIdx.x;   // 0 .. 3*H*C-1
    const int sel = idx >> 16;                        // H*C == 65536
    const int rem = idx & 65535;
    const int h   = rem >> 10;                        // / C
    const int k   = rem & (CC - 1);
    const float* W = (sel == 0) ? Wk : (sel == 1) ? Wq : Wv;
    const float  v = W[(size_t)k * HH + h] * ((sel == 1) ? SCALE : 1.0f);
    Wt[idx] = (_Float16)v;
}

// ---------------------------------------------------------------------------
// Kernel 1: QKV projection.  grid = (ROWS/64, 3), block = 128 (4 waves).
// Each WG computes a 64x64 tile of one projection; K=1024 in 32-chunks.
// sel==2 writes the V projection transposed per batch: fVt[b][d][t].
// ---------------------------------------------------------------------------
__global__ __launch_bounds__(128) void head_qkv_proj(
    const float* __restrict__ tokens, const _Float16* __restrict__ Wt,
    _Float16* __restrict__ fQ, _Float16* __restrict__ fK,
    _Float16* __restrict__ fVt)
{
    __shared__ _Float16 At[64][40];   // tokens tile, 64 rows x 32 K (pad->40)
    __shared__ _Float16 Bt[64][40];   // W^T tile: Bt[h][k], contiguous copy

    const int tid  = threadIdx.x;
    const int wave = tid >> 5;
    const int lane = tid & 31;
    const int lm   = lane & 15;       // m (A) / n (B,C) role
    const int lh   = lane >> 4;       // k-half (A,B) / m-half (C) role
    const int rows0 = blockIdx.x * 64;
    const int sel   = blockIdx.y;
    const _Float16* WtS = Wt + (size_t)sel * HH * CC;

    v8f acc[4] = {v8f{}, v8f{}, v8f{}, v8f{}};

    for (int kk = 0; kk < CC; kk += 32) {
        __syncthreads();
        {   // tokens[rows0..+63][kk..+31] (f32) -> At (f16), vector ds stores
            const int r  = tid >> 1;
            const int c0 = (tid & 1) * 16;
            const float* src = tokens + (size_t)(rows0 + r) * CC + kk + c0;
            v8h p0, p1;
            #pragma unroll
            for (int i = 0; i < 8; ++i) {
                p0[i] = (_Float16)src[i];
                p1[i] = (_Float16)src[8 + i];
            }
            *(v8h*)&At[r][c0]     = p0;
            *(v8h*)&At[r][c0 + 8] = p1;
        }
        {   // W^T[h][kk..+31] (f16) -> Bt, contiguous (async DMA capable)
            const int h  = tid >> 1;
            const int c0 = (tid & 1) * 16;
            const _Float16* src = WtS + (size_t)h * CC + kk + c0;
            g2l16(src,     &Bt[h][c0]);
            g2l16(src + 8, &Bt[h][c0 + 8]);
        }
        async_wait();
        __syncthreads();

        // A-frag: row = wave*16 + lm, K = 0..31 of this chunk
        v16h a;
        {
            const v8h lo = *(const v8h*)&At[wave * 16 + lm][lh * 16];
            const v8h hi = *(const v8h*)&At[wave * 16 + lm][lh * 16 + 8];
            #pragma unroll
            for (int i = 0; i < 8; ++i) { a[i] = lo[i]; a[8 + i] = hi[i]; }
        }
        #pragma unroll
        for (int nb = 0; nb < 4; ++nb) {
            const v16h b = *(const v16h*)&Bt[nb * 16 + lm][lh * 16];
            acc[nb] = __builtin_amdgcn_wmma_f32_16x16x32_f16(
                false, a, false, b, (short)0, acc[nb], false, false);
        }
    }

    // epilogue: C-layout (lane = mh*16 + n, vgpr r -> row mh*8+r)
    if (sel == 2) {
        // transposed store: fVt[b][d][t]; r walks consecutive t -> b128 store
        const int bidx = rows0 / TT;                      // uniform per WG
        const int t0   = (rows0 - bidx * TT) + wave * 16 + lh * 8;
        _Float16* vb = fVt + (size_t)bidx * HH * TT;
        #pragma unroll
        for (int nb = 0; nb < 4; ++nb) {
            v8h pk;
            #pragma unroll
            for (int r = 0; r < 8; ++r) pk[r] = (_Float16)acc[nb][r];
            *(v8h*)(vb + (size_t)(nb * 16 + lm) * TT + t0) = pk;
        }
    } else {
        _Float16* dst = (sel == 0) ? fQ : fK;
        #pragma unroll
        for (int nb = 0; nb < 4; ++nb)
            #pragma unroll
            for (int r = 0; r < 8; ++r) {
                const int row = rows0 + wave * 16 + lh * 8 + r;
                dst[(size_t)row * HH + nb * 16 + lm] = (_Float16)acc[nb][r];
            }
    }
}

// ---------------------------------------------------------------------------
// Kernel 2: causal flash attention.  grid = (T/64, B), block = 128 (4 waves).
// WG owns 64 query rows; each wave owns 16 rows x full head dim.
// ---------------------------------------------------------------------------
__global__ __launch_bounds__(128) void head_flash_attn(
    const _Float16* __restrict__ fQ, const _Float16* __restrict__ fK,
    const _Float16* __restrict__ fVt, float* __restrict__ out)
{
    __shared__ _Float16 Ks[64][72];       // fK rows (row-major), padded
    __shared__ _Float16 Vt[64][72];       // V^T tile: Vt[d][t]
    __shared__ _Float16 Pst[4][16][72];   // per-wave P staging (C -> A layout)

    const int tid  = threadIdx.x;
    const int wave = tid >> 5;
    const int lane = tid & 31;
    const int lm   = lane & 15;           // m (A) / n (B,C)
    const int lh   = lane >> 4;           // k-half (A,B) / m-half (C)
    const int i0   = blockIdx.x * 64;
    const size_t base  = (size_t)blockIdx.y * TT * HH;   // fQ/fK/out base
    const size_t vbase = (size_t)blockIdx.y * HH * TT;   // fVt base

    // Q fragments: rows i0 + wave*16 + lm, K=64 split into two 32-chunks
    v16h aq[2];
    {
        const _Float16* qrow = fQ + base + (size_t)(i0 + wave * 16 + lm) * HH;
        #pragma unroll
        for (int c = 0; c < 2; ++c) {
            const v8h lo = *(const v8h*)&qrow[c * 32 + lh * 16];
            const v8h hi = *(const v8h*)&qrow[c * 32 + lh * 16 + 8];
            #pragma unroll
            for (int i = 0; i < 8; ++i) { aq[c][i] = lo[i]; aq[c][8 + i] = hi[i]; }
        }
    }

    float rowmax[8], rowsum[8];
    v8f O[4] = {v8f{}, v8f{}, v8f{}, v8f{}};
    #pragma unroll
    for (int r = 0; r < 8; ++r) { rowmax[r] = -__builtin_inff(); rowsum[r] = 0.f; }

    for (int j0 = 0; j0 <= i0; j0 += 64) {
        __syncthreads();
        {   // stage K block and V^T block: contiguous global -> LDS DMA
            const int r  = tid >> 1;             // row (K) / head-dim (Vt)
            const int c0 = (tid & 1) * 32;
            const _Float16* ksrc = fK  + base  + (size_t)(j0 + r) * HH + c0;
            const _Float16* vsrc = fVt + vbase + (size_t)r * TT + j0 + c0;
            #pragma unroll
            for (int i = 0; i < 4; ++i) {
                g2l16(ksrc + i * 8, &Ks[r][c0 + i * 8]);
                g2l16(vsrc + i * 8, &Vt[r][c0 + i * 8]);
            }
        }
        if (j0 + 64 <= i0) {  // gfx1250 global_prefetch_b8 for next block
            __builtin_prefetch(fK + base + (size_t)(j0 + 64) * HH + tid * 32, 0, 1);
        }
        async_wait();
        __syncthreads();

        // S tile (16 x 64 per wave):  S = Q K'^T
        v8f s[4];
        #pragma unroll
        for (int nb = 0; nb < 4; ++nb) {
            v8f cacc = v8f{};
            #pragma unroll
            for (int c = 0; c < 2; ++c) {
                const v16h bk = *(const v16h*)&Ks[nb * 16 + lm][c * 32 + lh * 16];
                cacc = __builtin_amdgcn_wmma_f32_16x16x32_f16(
                    false, aq[c], false, bk, (short)0, cacc, false, false);
            }
            s[nb] = cacc;
        }

        // causal mask + online softmax (row state per vgpr r, per 16-lane half)
        #pragma unroll
        for (int r = 0; r < 8; ++r) {
            const int gi = i0 + wave * 16 + lh * 8 + r;
            float mx = rowmax[r];
            #pragma unroll
            for (int nb = 0; nb < 4; ++nb) {
                const int j = j0 + nb * 16 + lm;
                if (j > gi) s[nb][r] = -__builtin_inff();
                mx = fmaxf(mx, s[nb][r]);
            }
            #pragma unroll
            for (int off = 8; off >= 1; off >>= 1)
                mx = fmaxf(mx, __shfl_xor(mx, off, 32));
            const float alpha = __expf(rowmax[r] - mx);
            rowmax[r] = mx;
            float lsum = 0.f;
            #pragma unroll
            for (int nb = 0; nb < 4; ++nb) {
                const float p = __expf(s[nb][r] - mx);
                s[nb][r] = p;
                lsum += p;
                O[nb][r] *= alpha;
            }
            #pragma unroll
            for (int off = 8; off >= 1; off >>= 1)
                lsum += __shfl_xor(lsum, off, 32);
            rowsum[r] = rowsum[r] * alpha + lsum;
        }

        // stage P (f16) through LDS: C-layout -> A-layout
        #pragma unroll
        for (int nb = 0; nb < 4; ++nb)
            #pragma unroll
            for (int r = 0; r < 8; ++r)
                Pst[wave][lh * 8 + r][nb * 16 + lm] = (_Float16)s[nb][r];
        __syncthreads();

        // O += P @ V  (kk = key index 0..63 in two 32-chunks)
        #pragma unroll
        for (int c = 0; c < 2; ++c) {
            v16h ap;
            const v8h lo = *(const v8h*)&Pst[wave][lm][c * 32 + lh * 16];
            const v8h hi = *(const v8h*)&Pst[wave][lm][c * 32 + lh * 16 + 8];
            #pragma unroll
            for (int i = 0; i < 8; ++i) { ap[i] = lo[i]; ap[8 + i] = hi[i]; }
            #pragma unroll
            for (int nb = 0; nb < 4; ++nb) {
                const v16h bv = *(const v16h*)&Vt[nb * 16 + lm][c * 32 + lh * 16];
                O[nb] = __builtin_amdgcn_wmma_f32_16x16x32_f16(
                    false, ap, false, bv, (short)0, O[nb], false, false);
            }
        }
    }

    // normalize and store fp32 output
    #pragma unroll
    for (int nb = 0; nb < 4; ++nb)
        #pragma unroll
        for (int r = 0; r < 8; ++r) {
            const int gi = i0 + wave * 16 + lh * 8 + r;
            out[base + (size_t)gi * HH + nb * 16 + lm] = O[nb][r] / rowsum[r];
        }
}

// ---------------------------------------------------------------------------
extern "C" void kernel_launch(void* const* d_in, const int* in_sizes, int n_in,
                              void* d_out, int out_size, void* d_ws, size_t ws_size,
                              hipStream_t stream) {
    const float* tokens = (const float*)d_in[0];
    const float* Wk     = (const float*)d_in[1];
    const float* Wq     = (const float*)d_in[2];
    const float* Wv     = (const float*)d_in[3];
    float*       out    = (float*)d_out;

    _Float16* fQ  = (_Float16*)d_ws;                // flash-Q  = X@Wk
    _Float16* fK  = fQ  + (size_t)ROWS * HH;        // flash-K  = X@(Wq*scale)
    _Float16* fVt = fK  + (size_t)ROWS * HH;        // flash-V^T per batch
    _Float16* Wt  = fVt + (size_t)ROWS * HH;        // packed W^T (f16), 3 mats

    head_pack_w<<<dim3(3 * HH * CC / 256), 256, 0, stream>>>(Wk, Wq, Wv, Wt);
    head_qkv_proj<<<dim3(ROWS / 64, 3), 128, 0, stream>>>(
        tokens, Wt, fQ, fK, fVt);
    head_flash_attn<<<dim3(TT / 64, NB), 128, 0, stream>>>(fQ, fK, fVt, out);
}
